// DimensionMoE_56229711839482
// MI455X (gfx1250) — matrix operand order repaired and checked by reading, not compiled, not run on hardware
//
#include <hip/hip_runtime.h>
#include <hip/hip_bf16.h>
#include <math.h>

// Problem constants (from reference)
#define BB 256
#define TT 128
#define DD 1024
#define EE 64
#define HH 128
#define TOPK 4

typedef __attribute__((ext_vector_type(16))) __bf16 v16bf;
typedef __attribute__((ext_vector_type(8)))  float  v8f;

typedef int i32x4 __attribute__((vector_size(16)));
#define GLOBAL_AS __attribute__((address_space(1)))
#define LDS_AS    __attribute__((address_space(3)))

#if __has_builtin(__builtin_amdgcn_global_load_async_to_lds_b128)
#define HAVE_ASYNC_LDS 1
#else
#define HAVE_ASYNC_LDS 0
#endif

// ---------- d_out layout (floats) ----------
// [0,16384)        dim_scores  [B,E]
// [16384,32768)    gate_weights[B,E]
// [32768,33792)    topk_indices[B,4] (as float)
// [33792,34816)    topk_weights[B,4]
// [34816]          load_balance_loss
// [34817]          sparsity_loss
#define OUT_GATE   16384
#define OUT_TIDX   32768
#define OUT_TW     33792
#define OUT_LBL    34816
#define OUT_SPL    34817

// ---------- workspace layout ----------
#define WS_STATS_OFF   (BB*DD)
#define WS_BF16_BYTE   ((BB*DD + 144) * 4)

// round-to-nearest-even f32 -> bf16 pair packed in one dword
__device__ __forceinline__ unsigned pack_bf16(float a, float b) {
    unsigned ua = __float_as_uint(a);
    unsigned ub = __float_as_uint(b);
    ua = (ua + 0x7FFFu + ((ua >> 16) & 1u)) >> 16;
    ub = (ub + 0x7FFFu + ((ub >> 16) & 1u)) >> 16;
    return ua | (ub << 16);
}

// branchless exact-ish GELU: 0.5*x*(1+erf(x/sqrt(2))), A&S 7.1.26 erf (|err|<1.5e-7)
__device__ __forceinline__ float gelu_erf(float x) {
    float z  = x * 0.70710678118654752f;
    float az = fabsf(z);
    float t  = __builtin_amdgcn_rcpf(1.0f + 0.3275911f * az);
    float poly = t * (0.254829592f +
                 t * (-0.284496736f +
                 t * (1.421413741f +
                 t * (-1.453152027f +
                 t *  1.061405429f))));
    float er = 1.0f - poly * __expf(-az * az);
    er = copysignf(er, z);
    return 0.5f * x * (1.0f + er);
}

// ===================== kernel 0: zero stats =====================
__global__ __launch_bounds__(256) void zero_stats_kernel(float* stats) {
    int t = threadIdx.x;
    if (t < 129) stats[t] = 0.0f;
}

// ===================== kernel 1: time pooling (float4 vectorized) =====================
__global__ __launch_bounds__(256) void pool_kernel(const float* __restrict__ x,
                                                   float* __restrict__ xpool,
                                                   unsigned* __restrict__ xpool_bf_u32) {
    int idx4 = blockIdx.x * 256 + threadIdx.x;    // [0, B*D/4)
    int b  = idx4 >> 8;                           // / (D/4)
    int d4 = idx4 & 255;
    const float4* p = (const float4*)(x + (size_t)b * TT * DD) + d4;
    float4 s = make_float4(0.f, 0.f, 0.f, 0.f);
#pragma unroll 4
    for (int t = 0; t < TT; ++t) {
        float4 v = p[(size_t)t * (DD / 4)];
        s.x += v.x; s.y += v.y; s.z += v.z; s.w += v.w;
    }
    const float r = 1.0f / (float)TT;
    float4 avg = make_float4(s.x * r, s.y * r, s.z * r, s.w * r);
    ((float4*)xpool)[idx4] = avg;
    uint2 packed = make_uint2(pack_bf16(avg.x, avg.y), pack_bf16(avg.z, avg.w));
    ((uint2*)xpool_bf_u32)[idx4] = packed;
}

// ===================== kernel 2: router =====================
__global__ __launch_bounds__(64) void router_kernel(const float* __restrict__ xpool,
                                                    const float* __restrict__ Wg,
                                                    const float* __restrict__ bg,
                                                    float* __restrict__ out,
                                                    float* __restrict__ stats) {
    __shared__ float xrow[DD];
    __shared__ float sh[EE];
    __shared__ float sv[EE];
    __shared__ float tw[TOPK];
    __shared__ int   ti[TOPK];
    __shared__ float mx, sm;

    const int b = blockIdx.x;
    const int e = threadIdx.x;

    for (int i = e; i < DD; i += 64) xrow[i] = xpool[(size_t)b * DD + i];
    __syncthreads();

    float acc = 0.0f;
#pragma unroll 4
    for (int d = 0; d < DD; ++d) acc += xrow[d] * Wg[(size_t)d * EE + e];
    float logit = acc + bg[e];
    sh[e] = logit;
    __syncthreads();

    if (e == 0) {
        float m = sh[0];
        for (int i = 1; i < EE; ++i) m = fmaxf(m, sh[i]);
        mx = m;
    }
    __syncthreads();
    float g = __expf(logit - mx);
    sh[e] = g;
    __syncthreads();
    if (e == 0) {
        float s = 0.0f;
        for (int i = 0; i < EE; ++i) s += sh[i];
        sm = s;
    }
    __syncthreads();
    float gate = g / sm;
    out[OUT_GATE + b * EE + e] = gate;
    atomicAdd(&stats[e], gate * (1.0f / (float)BB));   // importance

    sv[e] = gate;
    __syncthreads();

    if (e == 0) {
        for (int k = 0; k < TOPK; ++k) {
            int bi = 0; float bv = sv[0];
            for (int i = 1; i < EE; ++i) { if (sv[i] > bv) { bv = sv[i]; bi = i; } }
            tw[k] = bv; ti[k] = bi; sv[bi] = -1.0f;
        }
        float s4 = 0.0f;
        for (int k = 0; k < TOPK; ++k) {
            s4 += tw[k];
            out[OUT_TIDX + b * TOPK + k] = (float)ti[k];
            out[OUT_TW   + b * TOPK + k] = tw[k];
            atomicAdd(&stats[EE + ti[k]], 1.0f / (float)BB);   // load
        }
        atomicAdd(&stats[2 * EE], s4 * (1.0f / (float)BB));    // mean topk sum
    }
}

// ===================== kernel 3: fused expert GEMM (WMMA) =====================
// grid: 256 blocks = E(64) x mtiles(4); block = 256 threads (8 waves, wave32)
// C[64 x 128] = Xp[mt*64 : +64, :] @ W1[e]  (K=1024, bf16 WMMA, f32 acc)
// Epilogue: +b1, GELU, dot with W2 over H, +b2, * gate -> dim_scores
__global__ __launch_bounds__(256) void expert_kernel(const __bf16* __restrict__ xpool_bf,
                                                     const float*  __restrict__ W1,
                                                     const float*  __restrict__ b1,
                                                     const float*  __restrict__ W2,
                                                     const float*  __restrict__ b2,
                                                     const float*  __restrict__ gate,
                                                     float* __restrict__ dim_scores) {
    __shared__ __align__(16) __bf16   lA[64 * 32];    // A tile, [m][k]
    __shared__ __align__(16) unsigned lBu[HH * 16];   // W1 tile transposed [h][k], k-pairs packed
    __shared__ float rowsum[64];
    __bf16* lB = (__bf16*)lBu;

    const int tid  = threadIdx.x;
    const int lane = tid & 31;
    const int wid  = tid >> 5;
    const int wm   = wid & 3;        // 16-row strip within 64
    const int wn   = wid >> 2;       // 64-col strip within 128 (0/1)

    const int e  = blockIdx.x >> 2;  // expert
    const int mt = blockIdx.x & 3;   // m tile

    if (tid < 64) rowsum[tid] = 0.0f;

    // A staging: each thread -> one b128 (8 bf16)
    const int ar   = tid >> 2;       // row 0..63
    const int aseg = tid & 3;        // 8-col segment
    // B staging: each thread -> k-pair x 8 h values
    const int kp = tid >> 4;         // 0..15 -> k = 2*kp, 2*kp+1
    const int h0 = (tid & 15) * 8;   // 0..120

    const float* wbase = W1 + (size_t)e * DD * HH;

    v8f acc[4] = {};

    const int half = lane >> 4;
    const int lm   = lane & 15;

    for (int k0 = 0; k0 < DD; k0 += 32) {
        // ---- stage A tile (async DMA to LDS when available) ----
        const __bf16* asrc = xpool_bf + (size_t)(mt * 64 + ar) * DD + k0 + aseg * 8;
#if HAVE_ASYNC_LDS
        __builtin_amdgcn_global_load_async_to_lds_b128(
            (GLOBAL_AS i32x4*)asrc,
            (LDS_AS i32x4*)(((uint4*)lA) + tid),
            0, 0);
#else
        ((uint4*)lA)[tid] = *((const uint4*)asrc);
#endif

        // ---- stage W1 tile: 2 k-rows x 8 h, pack bf16 pairs, 8x ds_store_b32 ----
        const float* wrow0 = wbase + (size_t)(k0 + 2 * kp) * HH + h0;
        const float* wrow1 = wrow0 + HH;
        float4 a0 = ((const float4*)wrow0)[0];
        float4 a1 = ((const float4*)wrow0)[1];
        float4 c0 = ((const float4*)wrow1)[0];
        float4 c1 = ((const float4*)wrow1)[1];
        if (k0 + 32 < DD) __builtin_prefetch(wrow0 + 32 * HH, 0, 0);
        unsigned* dst = &lBu[(size_t)h0 * 16 + kp];   // stride 16 dwords per h
        dst[0 * 16] = pack_bf16(a0.x, c0.x);
        dst[1 * 16] = pack_bf16(a0.y, c0.y);
        dst[2 * 16] = pack_bf16(a0.z, c0.z);
        dst[3 * 16] = pack_bf16(a0.w, c0.w);
        dst[4 * 16] = pack_bf16(a1.x, c1.x);
        dst[5 * 16] = pack_bf16(a1.y, c1.y);
        dst[6 * 16] = pack_bf16(a1.z, c1.z);
        dst[7 * 16] = pack_bf16(a1.w, c1.w);

#if HAVE_ASYNC_LDS && __has_builtin(__builtin_amdgcn_s_wait_asynccnt)
        __builtin_amdgcn_s_wait_asynccnt(0);
#endif
        __syncthreads();

        // ---- A fragment (16x32 bf16, ISA layout) ----
        const __bf16* arow = &lA[(wm * 16 + lm) * 32 + half * 8];
        v16bf afrag;
#pragma unroll
        for (int i = 0; i < 8; ++i) {
            afrag[i]     = arow[i];
            afrag[8 + i] = arow[16 + i];
        }

        // ---- 4 B fragments + WMMAs ----
#pragma unroll
        for (int j = 0; j < 4; ++j) {
            const int col = wn * 64 + j * 16 + lm;
            const __bf16* bcol = &lB[col * 32 + half * 8];
            v16bf bfrag;
#pragma unroll
            for (int i = 0; i < 16; ++i) bfrag[i] = bcol[i < 8 ? i : i + 8];

            acc[j] = __builtin_amdgcn_wmma_f32_16x16x32_bf16(
                false, afrag, false, bfrag, (short)0, acc[j], false, false);
        }
        __syncthreads();
    }

    // ---- epilogue: +b1, branchless GELU, *W2, reduce over H ----
#pragma unroll
    for (int v = 0; v < 8; ++v) {
        const int rloc = wm * 16 + half * 8 + v;
        float partial = 0.0f;
#pragma unroll
        for (int j = 0; j < 4; ++j) {
            const int col = wn * 64 + j * 16 + lm;
            float xv = acc[j][v] + b1[e * HH + col];
            partial += gelu_erf(xv) * W2[e * HH + col];
        }
        atomicAdd(&rowsum[rloc], partial);
    }
    __syncthreads();

    if (tid < 64) {
        const int gb = mt * 64 + tid;
        float val = rowsum[tid] + b2[e];
        dim_scores[gb * EE + e] = val * gate[gb * EE + e];
    }
}

// ===================== kernel 4: losses =====================
__global__ __launch_bounds__(64) void loss_kernel(const float* __restrict__ stats,
                                                  float* __restrict__ out) {
    __shared__ float sh[EE];
    int e = threadIdx.x;
    sh[e] = stats[e] * stats[EE + e];
    __syncthreads();
    if (e == 0) {
        float s = 0.0f;
        for (int i = 0; i < EE; ++i) s += sh[i];
        out[OUT_LBL] = (float)EE * s;
        out[OUT_SPL] = 1.0f - stats[2 * EE];
    }
}

// ===================== launcher =====================
extern "C" void kernel_launch(void* const* d_in, const int* in_sizes, int n_in,
                              void* d_out, int out_size, void* d_ws, size_t ws_size,
                              hipStream_t stream) {
    const float* x  = (const float*)d_in[0];
    const float* Wg = (const float*)d_in[1];
    const float* bg = (const float*)d_in[2];
    const float* W1 = (const float*)d_in[3];
    const float* b1 = (const float*)d_in[4];
    const float* W2 = (const float*)d_in[5];
    const float* b2 = (const float*)d_in[6];

    float*    out      = (float*)d_out;
    float*    ws_f     = (float*)d_ws;
    float*    xpool    = ws_f;
    float*    stats    = ws_f + WS_STATS_OFF;
    unsigned* xpool_bu = (unsigned*)((char*)d_ws + WS_BF16_BYTE);
    __bf16*   xpool_bf = (__bf16*)xpool_bu;

    zero_stats_kernel<<<1, 256, 0, stream>>>(stats);
    pool_kernel<<<(BB * DD / 4) / 256, 256, 0, stream>>>(x, xpool, xpool_bu);
    router_kernel<<<BB, 64, 0, stream>>>(xpool, Wg, bg, out, stats);
    expert_kernel<<<EE * 4, 256, 0, stream>>>(xpool_bf, W1, b1, W2, b2,
                                              out + OUT_GATE, out);
    loss_kernel<<<1, 64, 0, stream>>>(stats, out);
}
